// ViewportFeatureEncoder_39659728011311
// MI455X (gfx1250) — matrix operand order repaired and checked by reading, not compiled, not run on hardware
//
#include <hip/hip_runtime.h>
#include <hip/hip_bf16.h>

// ---------------------------------------------------------------------------
// y[b,t,o] = leaky_relu( sum_i x[b,t,i] * W[o,i] + b[o] ),  slope = 0.01
// GEMM view: M = 64*2048 = 131072, K = 256, N = 1024, fp32 in/out.
// Strategy: fp32 -> bf16 in LDS once per tile, V_WMMA_F32_16X16X32_BF16,
// f32 accumulate, fused bias + leaky-ReLU epilogue, NT stores for Y.
// ---------------------------------------------------------------------------

typedef __attribute__((ext_vector_type(16))) __bf16 v16bf;
typedef __attribute__((ext_vector_type(8)))  __bf16 v8bf;
typedef __attribute__((ext_vector_type(8)))  float  v8f;

#define IN_CH     256
#define OUT_CH    1024
#define NEG_SLOPE 0.01f

#define BM 32      // rows per block
#define BN 256     // cols per block
#define BK 32      // K slice per stage
#define LDS_PAD 8  // +16B per row: spreads LDS banks, keeps 16B alignment

__device__ __forceinline__ v16bf cat16(v8bf a, v8bf b) {
    return __builtin_shufflevector(a, b, 0, 1, 2, 3, 4, 5, 6, 7,
                                         8, 9, 10, 11, 12, 13, 14, 15);
}

__global__ __launch_bounds__(256)
void vfe_linear_lrelu_wmma(const float* __restrict__ X,
                           const float* __restrict__ W,
                           const float* __restrict__ bias,
                           float* __restrict__ Y,
                           int M) {
    __shared__ __bf16 As[BM][BK + LDS_PAD];   //  2.5 KB
    __shared__ __bf16 Bs[BN][BK + LDS_PAD];   // 20.0 KB

    const int tid  = threadIdx.x;
    const int lane = tid & 31;          // wave32
    const int wave = tid >> 5;          // 0..7
    const int mh   = wave >> 2;         // wave's M half: 0..1  (16 rows each)
    const int nq   = wave & 3;          // wave's N quarter: 0..3 (64 cols each)

    const int m0 = blockIdx.y * BM;
    const int n0 = blockIdx.x * BN;
    if (m0 >= M) return;

    const int half = lane >> 4;         // 0: lanes 0-15, 1: lanes 16-31
    const int l16  = lane & 15;

    v8f acc[4] = {};                    // 4 tiles of 16x16 f32 along N

    for (int kk = 0; kk < IN_CH; kk += BK) {
        __syncthreads();                // previous iter consumers done

        // ---- stage A tile (32 x 32 fp32 -> bf16), 4 floats / thread ----
        {
            const int row = tid >> 3;           // 0..31
            const int col = (tid & 7) * 4;      // 0,4,...,28
            const float4 a = *reinterpret_cast<const float4*>(
                X + (size_t)(m0 + row) * IN_CH + kk + col);
            As[row][col + 0] = (__bf16)a.x;
            As[row][col + 1] = (__bf16)a.y;
            As[row][col + 2] = (__bf16)a.z;
            As[row][col + 3] = (__bf16)a.w;
        }

        // ---- stage B tile: thread t loads W[n0+t][kk .. kk+31] ----
        {
            const float* wp = W + (size_t)(n0 + tid) * IN_CH + kk;
#pragma unroll
            for (int c = 0; c < BK; c += 4) {
                const float4 w4 = *reinterpret_cast<const float4*>(wp + c);
                Bs[tid][c + 0] = (__bf16)w4.x;
                Bs[tid][c + 1] = (__bf16)w4.y;
                Bs[tid][c + 2] = (__bf16)w4.z;
                Bs[tid][c + 3] = (__bf16)w4.w;
            }
        }

        __syncthreads();

        // ---- A fragment (16x32 bf16), ISA layout:
        //      lanes 0-15: row l16, K 0-7 & 16-23 ; lanes 16-31: K 8-15 & 24-31
        const int arow = mh * 16 + l16;
        const v8bf a0 = *reinterpret_cast<const v8bf*>(&As[arow][half ? 8 : 0]);
        const v8bf a1 = *reinterpret_cast<const v8bf*>(&As[arow][half ? 24 : 16]);
        const v16bf afrag = cat16(a0, a1);

        // ---- 4 B fragments + WMMA, tiles t cover cols nq*64 + t*16 ----
#pragma unroll
        for (int t = 0; t < 4; ++t) {
            const int brow = nq * 64 + t * 16 + l16;   // B column index = W row
            // lanes 0-15 hold K 0-7 & 16-23, lanes 16-31 hold K 8-15 & 24-31 of column brow
            const v8bf b0 = *reinterpret_cast<const v8bf*>(&Bs[brow][half * 8]);
            const v8bf b1 = *reinterpret_cast<const v8bf*>(&Bs[brow][16 + half * 8]);
            const v16bf bfrag = cat16(b0, b1);

            acc[t] = __builtin_amdgcn_wmma_f32_16x16x32_bf16(
                /*neg_a=*/false, afrag, /*neg_b=*/false, bfrag,
                /*c_mod=*/(short)0, acc[t],
                /*reuse_a=*/false, /*reuse_b=*/false);
        }
    }

    // ---- epilogue: bias + leaky ReLU + NT store ----
    // C/D layout: VGPR r -> lanes 0-15: (M=r, N=l16); lanes 16-31: (M=r+8, N=l16)
    const int gm_base = m0 + mh * 16 + half * 8;
#pragma unroll
    for (int t = 0; t < 4; ++t) {
        const int gn = n0 + nq * 64 + t * 16 + l16;
        const float bb = bias[gn];
#pragma unroll
        for (int r = 0; r < 8; ++r) {
            float v = acc[t][r] + bb;
            v = (v >= 0.0f) ? v : NEG_SLOPE * v;
            __builtin_nontemporal_store(
                v, Y + (size_t)(gm_base + r) * OUT_CH + gn);
        }
    }
}

extern "C" void kernel_launch(void* const* d_in, const int* in_sizes, int n_in,
                              void* d_out, int out_size, void* d_ws, size_t ws_size,
                              hipStream_t stream) {
    const float* X = (const float*)d_in[0];   // (64, 2048, 256) fp32
    const float* W = (const float*)d_in[1];   // (1024, 256) fp32
    const float* B = (const float*)d_in[2];   // (1024,) fp32
    float* Y = (float*)d_out;                 // (64, 2048, 1024) fp32

    const int M = in_sizes[0] / IN_CH;        // 131072
    dim3 grid(OUT_CH / BN, (M + BM - 1) / BM);
    dim3 block(256);
    vfe_linear_lrelu_wmma<<<grid, block, 0, stream>>>(X, W, B, Y, M);
}
